// Codebook_72138270704376
// MI455X (gfx1250) — matrix operand order repaired and checked by reading, not compiled, not run on hardware
//
#include <hip/hip_runtime.h>
#include <stdint.h>

// Problem constants (B=65536, N=512, D=512 from the reference).
#define B_SZ 65536
#define N_SZ 512
#define D_SZ 512

typedef __attribute__((ext_vector_type(16))) __bf16 v16bf;
typedef __attribute__((ext_vector_type(8)))  float  v8f;

// LDS chunk: 16 codebook rows x 512 bf16, padded row stride (32B aligned,
// bank-staggered: 1056B = 264 dwords -> +8 banks per row).
#define ROW_BYTES 1056u
#define BUF_BYTES (16u * ROW_BYTES)   // 16896, multiple of 32

// ---------------------------------------------------------------------------
// Kernel 1: zero the one-hot region (128MB). 8192 blocks x 256 thr x 4 float4.
// ---------------------------------------------------------------------------
__global__ __launch_bounds__(256) void fill_zero_kernel(float4* __restrict__ out) {
  size_t i = ((size_t)blockIdx.x * blockDim.x + threadIdx.x) * 4;
  float4 z = make_float4(0.f, 0.f, 0.f, 0.f);
#pragma unroll
  for (int k = 0; k < 4; ++k) out[i + k] = z;
}

// ---------------------------------------------------------------------------
// Kernel 2: codebook f32 -> bf16 (row-major [512][512] in d_ws) + ||c||^2.
// One wave per codebook row; each lane converts 16 consecutive elements.
// ---------------------------------------------------------------------------
__global__ __launch_bounds__(32) void cvt_codebook_kernel(
    const float* __restrict__ cb, __bf16* __restrict__ wcb,
    float* __restrict__ csq) {
  const int row  = blockIdx.x;
  const int lane = threadIdx.x;  // 0..31
  const float4* src = (const float4*)(cb + (size_t)row * D_SZ);
  float s = 0.f;
  v16bf v;
#pragma unroll
  for (int i = 0; i < 4; ++i) {
    float4 x = src[lane * 4 + i];
    s += x.x * x.x + x.y * x.y + x.z * x.z + x.w * x.w;
    v[i * 4 + 0] = (__bf16)x.x;
    v[i * 4 + 1] = (__bf16)x.y;
    v[i * 4 + 2] = (__bf16)x.z;
    v[i * 4 + 3] = (__bf16)x.w;
  }
  *(v16bf*)(wcb + (size_t)row * D_SZ + lane * 16) = v;
#pragma unroll
  for (int m = 16; m >= 1; m >>= 1) s += __shfl_xor(s, m, 32);
  if (lane == 0) csq[row] = s;
}

// ---------------------------------------------------------------------------
// Async global -> LDS copy, 16 bytes per issue (CDNA5 async path, ASYNCcnt).
// ---------------------------------------------------------------------------
__device__ __forceinline__ void async_cp16(uint32_t lds_off, uint64_t gaddr) {
  asm volatile("global_load_async_to_lds_b128 %0, %1, off"
               :
               : "v"(lds_off), "v"(gaddr)
               : "memory");
}

#define WAIT_ASYNC(n) asm volatile("s_wait_asynccnt " #n ::: "memory")

// Stage one 16-row x 512-K bf16 codebook chunk (16KB) into LDS buffer `buf`.
// 1024 x 16B chunks, 4 per thread (4 async issues -> ASYNCcnt += 4 per wave).
__device__ __forceinline__ void stage_chunk(const __bf16* wcb, int j, int buf,
                                            int tid) {
  uint64_t gbase = (uint64_t)wcb + (uint64_t)j * (16u * D_SZ * 2u);
  uint32_t lbase = (uint32_t)buf * BUF_BYTES;
#pragma unroll
  for (int k = 0; k < 4; ++k) {
    int c   = k * 256 + tid;   // 0..1023
    int row = c >> 6;          // 0..15
    int col = c & 63;          // 16B chunk within row
    async_cp16(lbase + (uint32_t)row * ROW_BYTES + (uint32_t)col * 16u,
               gbase + (uint64_t)(row * (D_SZ * 2) + col * 16));
  }
}

// ---------------------------------------------------------------------------
// Main kernel: 512 blocks x 256 threads (8 waves). Wave w owns rows
// [blockIdx.x*128 + w*16, +16). A tile kept in registers (16 bf16 frags),
// codebook chunks double-buffered in LDS, 16x16x32 bf16 WMMA, running argmin.
// ---------------------------------------------------------------------------
__global__ __launch_bounds__(256) void codebook_argmin_kernel(
    const float* __restrict__ zmat, const __bf16* __restrict__ wcb,
    const float* __restrict__ csq, float* __restrict__ onehot,
    float* __restrict__ outidx) {
  extern __shared__ char smem[];
  const int tid  = threadIdx.x;
  const int lane = tid & 31;
  const int wv   = tid >> 5;       // wave id 0..7
  const int l15  = lane & 15;
  const int half = lane >> 4;      // 0 or 1
  const int row0 = blockIdx.x * 128 + wv * 16;

  // Kick off async stage of column-tile 0 to overlap with A-tile load.
  stage_chunk(wcb, 0, 0, tid);

  // Load A fragments (contiguous K order, matching B):
  //   lanes 0-15 : row M=l15, K = kt*32 + {0..15}
  //   lanes 16-31: row M=l15, K = kt*32 + {16..31}
  const int arow = row0 + l15;
  const float4* zr = (const float4*)(zmat + (size_t)arow * D_SZ);
  v16bf afr[16];
#pragma unroll
  for (int kt = 0; kt < 16; ++kt) {
    float4 x0 = zr[kt * 8 + half * 4 + 0];
    float4 x1 = zr[kt * 8 + half * 4 + 1];
    float4 x2 = zr[kt * 8 + half * 4 + 2];
    float4 x3 = zr[kt * 8 + half * 4 + 3];
    v16bf a;
    a[0]  = (__bf16)x0.x; a[1]  = (__bf16)x0.y;
    a[2]  = (__bf16)x0.z; a[3]  = (__bf16)x0.w;
    a[4]  = (__bf16)x1.x; a[5]  = (__bf16)x1.y;
    a[6]  = (__bf16)x1.z; a[7]  = (__bf16)x1.w;
    a[8]  = (__bf16)x2.x; a[9]  = (__bf16)x2.y;
    a[10] = (__bf16)x2.z; a[11] = (__bf16)x2.w;
    a[12] = (__bf16)x3.x; a[13] = (__bf16)x3.y;
    a[14] = (__bf16)x3.z; a[15] = (__bf16)x3.w;
    afr[kt] = a;
  }

  // Running argmin of score = csq[n] - 2 * (z . c_n)  (z_sq + sqrt dropped:
  // constant per row / monotonic, argmin unchanged).
  float best[8];
  int   bidx[8];
#pragma unroll
  for (int r = 0; r < 8; ++r) { best[r] = 3.4e38f; bidx[r] = 0; }

  for (int j = 0; j < 32; ++j) {
    const int buf = j & 1;
    if (j < 31) {
      stage_chunk(wcb, j + 1, buf ^ 1, tid);  // prefetch next chunk
      WAIT_ASYNC(4);                          // chunk j landed (4 newer pending)
    } else {
      WAIT_ASYNC(0);
    }
    __syncthreads();

    const float cs = csq[j * 16 + l15];

    // B layout used here (contiguous K, matching A): lanes 0-15 hold K 0..15 (col N=l15),
    // lanes 16-31 hold K 16..31 -> 32 contiguous bytes per lane per frag.
    const char* bb = smem + (uint32_t)buf * BUF_BYTES +
                     (uint32_t)l15 * ROW_BYTES + (uint32_t)half * 32u;

    v8f acc = {0.f, 0.f, 0.f, 0.f, 0.f, 0.f, 0.f, 0.f};
#pragma unroll
    for (int kt = 0; kt < 16; ++kt) {
      v16bf bfrag = *(const v16bf*)(bb + kt * 64);
      acc = __builtin_amdgcn_wmma_f32_16x16x32_bf16(
          false, afr[kt], false, bfrag, (short)0, acc, false, false);
    }

    // C/D layout: vgpr r, lane -> row M = r + 8*half, col N = l15.
    const int n = j * 16 + l15;
#pragma unroll
    for (int r = 0; r < 8; ++r) {
      float s = cs - 2.0f * acc[r];
      if (s < best[r]) { best[r] = s; bidx[r] = n; }
    }
    __syncthreads();  // compute done before next stage overwrites this buffer
  }

  // Argmin across the 16 lanes of each half (ties -> smaller index, matching
  // jnp.argmin). xor masks <= 8 stay within each 16-lane half.
#pragma unroll
  for (int m = 8; m >= 1; m >>= 1) {
#pragma unroll
    for (int r = 0; r < 8; ++r) {
      float ov = __shfl_xor(best[r], m, 32);
      int   oi = __shfl_xor(bidx[r], m, 32);
      if (ov < best[r] || (ov == best[r] && oi < bidx[r])) {
        best[r] = ov;
        bidx[r] = oi;
      }
    }
  }

  if (l15 == 0) {  // lanes 0 (rows row0+0..7) and 16 (rows row0+8..15)
#pragma unroll
    for (int r = 0; r < 8; ++r) {
      int row = row0 + half * 8 + r;
      int n   = bidx[r];
      onehot[(size_t)row * N_SZ + n] = 1.0f;
      outidx[row] = (float)n;
    }
  }
}

// ---------------------------------------------------------------------------
// Host launcher.
// ---------------------------------------------------------------------------
extern "C" void kernel_launch(void* const* d_in, const int* in_sizes, int n_in,
                              void* d_out, int out_size, void* d_ws,
                              size_t ws_size, hipStream_t stream) {
  (void)in_sizes; (void)n_in; (void)out_size; (void)ws_size;
  const float* z  = (const float*)d_in[0];  // batch_z (B, D) f32
  const float* cb = (const float*)d_in[1];  // codebook (N, D) f32

  float* out    = (float*)d_out;            // one_hot (B, N) f32
  float* outidx = out + (size_t)B_SZ * N_SZ;  // indices (B,) as float

  __bf16* wcb = (__bf16*)d_ws;                                  // 512 KB
  float*  csq = (float*)((char*)d_ws + (size_t)N_SZ * D_SZ * 2);  // 2 KB

  // 1) zero one-hot region (d_out is poisoned before timing)
  fill_zero_kernel<<<8192, 256, 0, stream>>>((float4*)d_out);
  // 2) codebook -> bf16 + row norms into scratch
  cvt_codebook_kernel<<<N_SZ, 32, 0, stream>>>(cb, wcb, csq);
  // 3) WMMA GEMM + fused argmin + one-hot scatter
  codebook_argmin_kernel<<<B_SZ / 128, 256, 2 * BUF_BYTES, stream>>>(
      z, wcb, csq, out, outidx);
}
